// KalmanNetGRU_3238405341650
// MI455X (gfx1250) — matrix operand order repaired and checked by reading, not compiled, not run on hardware
//
#include <hip/hip_runtime.h>
#include <hip/hip_bf16.h>
#include <math.h>

// ---------------------------------------------------------------------------
// KalmanNet GRU — persistent-kernel CDNA5 implementation.
// B=1024, Q=128, D=64, H=256, 3H=768, W_OUT=32.
// 64 workgroups x 256 threads (8 waves). Each WG owns 16 batch rows and runs
// the whole encoder scan + rollout with state in LDS. All GEMMs use
// v_wmma_f32_16x16x32_bf16 with pre-swizzled bf16 weight panels in d_ws
// (L2-resident; ~1.8MB total vs 192MB L2). An opaque per-iteration offset
// keeps the B-fragment loads inside the recurrence loop (global_load_b128
// from L2) without losing the global address space (flat_load avoidance).
// ---------------------------------------------------------------------------

typedef __attribute__((ext_vector_type(16))) __bf16 v16bf;
typedef __attribute__((ext_vector_type(8)))  float  v8f;

#define B_SZ   1024
#define QLEN   128
#define DDIM   64
#define HDIM   256
#define H3     768
#define WOUT   32
#define FEATK  192
#define KROLL  320
#define ROWS   16
#define PI_F   3.14159265358979f

// bf16 weight-panel offsets in d_ws (dword units).
// Panel layout: [ntile][kstep][lane 0..31][vgpr 0..7] dwords;
// dword(lane,v) packs B[k][n],B[k+1][n] with n = ntile*16 + (lane&15),
// k = kstep*32 + (v<4 ? 2v+8*(lane>>4) : 16+2(v-4)+8*(lane>>4))  (B[k][n] == W[n][k]).
#define OFF_WF     0        // 16 ntiles x 6  ksteps
#define OFF_WIH    24576    // 48 x 8
#define OFF_WHH    122880   // 48 x 8
#define OFF_WG     221184   // 4  x 8
#define OFF_WIHR   229376   // 48 x 10
#define OFF_WHHR   352256   // 48 x 8
#define PANEL_TOTAL 450560

__device__ __forceinline__ unsigned pack2bf(float a, float b) {
  union { float f; unsigned u; } x, y; x.f = a; y.f = b;
  return (x.u >> 16) | (y.u & 0xFFFF0000u);
}
// Single-instruction v_rcp_f32 / v_rsq_f32 instead of IEEE divide/sqrt chains.
__device__ __forceinline__ float fast_rcp(float x) { return __builtin_amdgcn_rcpf(x); }
__device__ __forceinline__ float fast_rsq(float x) { return __builtin_amdgcn_rsqf(x); }
__device__ __forceinline__ float sigm(float x) {
  return fast_rcp(1.0f + __expf(-x));
}
// Branch-free tanh: tanh(x) = sign(x) * (1 - t)*rcp(1 + t), t = exp(-2|x|).
__device__ __forceinline__ float fast_tanh(float x) {
  float t = __expf(-2.0f * fabsf(x));
  return copysignf((1.0f - t) * fast_rcp(1.0f + t), x);
}
__device__ __forceinline__ v8f vzero() {
  v8f z;
#pragma unroll
  for (int i = 0; i < 8; ++i) z[i] = 0.0f;
  return z;
}
// Opaque zero offset: defeats LICM of panel loads across the recurrence loop
// while preserving the global address space of the base pointer.
__device__ __forceinline__ size_t opaque_zero() {
  unsigned long long u = 0;
  asm volatile("" : "+s"(u));
  return (size_t)u;
}

// --------------------------- weight prep ----------------------------------
__global__ __launch_bounds__(256) void prep_panels(
    const float* __restrict__ Wf, const float* __restrict__ Wih,
    const float* __restrict__ Whh, const float* __restrict__ Wg,
    const float* __restrict__ Wihr, const float* __restrict__ Whhr,
    unsigned* __restrict__ dst) {
  int d = blockIdx.x * blockDim.x + threadIdx.x;
  if (d >= PANEL_TOTAL) return;
  const float* src; int Kt; int base;
  if      (d < OFF_WIH)  { src = Wf;   Kt = FEATK; base = OFF_WF;   }
  else if (d < OFF_WHH)  { src = Wih;  Kt = HDIM;  base = OFF_WIH;  }
  else if (d < OFF_WG)   { src = Whh;  Kt = HDIM;  base = OFF_WHH;  }
  else if (d < OFF_WIHR) { src = Wg;   Kt = HDIM;  base = OFF_WG;   }
  else if (d < OFF_WHHR) { src = Wihr; Kt = KROLL; base = OFF_WIHR; }
  else                   { src = Whhr; Kt = HDIM;  base = OFF_WHHR; }
  int r   = d - base;
  int nks = Kt >> 5;
  int nt  = r / (nks * 256);
  int rem = r % (nks * 256);
  int ks  = rem >> 8;
  int q   = rem & 255;
  int l   = q >> 3, v = q & 7;
  int n   = nt * 16 + (l & 15);
  int k   = ks * 32 + ((v < 4) ? (2 * v + 8 * (l >> 4)) : (16 + 2 * (v - 4) + 8 * (l >> 4)));
  dst[d]  = pack2bf(src[n * Kt + k], src[n * Kt + k + 1]);
}

// --------------------------- fragment helpers ------------------------------
// A-fragment (16x32 bf16, M x K): lane l holds row M=l&15; half=l>>4 selects
// the K-interleave; vgpr v<4 -> k = 2v+8*half, v>=4 -> k = 16+2(v-4)+8*half.
// LDS staging: [kstep][lane][v] dwords, so each lane loads 8 contiguous dwords.
template <int NKS, typename F>
__device__ __forceinline__ void fill_afrag(unsigned* dst, int tid, F src) {
  int l = tid >> 3, v = tid & 7;
  int M = l & 15, half = l >> 4;
  int kp = (v < 4) ? (2 * v + 8 * half) : (16 + 2 * (v - 4) + 8 * half);
#pragma unroll
  for (int ks = 0; ks < NKS; ++ks) {
    int k = ks * 32 + kp;
    dst[ks * 256 + l * 8 + v] = pack2bf(src(M, k), src(M, k + 1));
  }
}

template <int NKS>
__device__ __forceinline__ v8f wmma_acc(const unsigned* afr,
                                        const unsigned* __restrict__ bfr,
                                        int lane, v8f acc) {
#pragma unroll
  for (int ks = 0; ks < NKS; ++ks) {
    v16bf a = *(const v16bf*)(afr + ks * 256 + lane * 8);
    v16bf b = *(const v16bf*)(bfr + ks * 256 + lane * 8);
    acc = __builtin_amdgcn_wmma_f32_16x16x32_bf16(
        /*neg_a=*/false, a, /*neg_b=*/false, b,
        /*c_mod=*/(short)0, acc, /*reuse_a=*/false, /*reuse_b=*/false);
  }
  return acc;
}

// GRU gates for this WG's 16 rows: gi = A_x @ W_i^T + b_i, gh = A_h @ W_h^T + b_h,
// r/z/n combine, h_new -> sOut (fp32 16x256). Wave w owns column tiles 2w, 2w+1
// of the 256-wide slice; the three gates share the column offset so r,z stay
// in registers for the n-gate combine.
template <int NKSI, int NKSH>
__device__ __forceinline__ void gru_block(
    const unsigned* axi, const unsigned* __restrict__ pw_i,
    const unsigned* axh, const unsigned* __restrict__ pw_h,
    const float* __restrict__ b_i, const float* __restrict__ b_h,
    const float* sHold, float* sOut, int wave, int lane) {
  int half = lane >> 4, nlo = lane & 15;
#pragma unroll
  for (int c = 0; c < 2; ++c) {
    int ct  = wave * 2 + c;
    int col = ct * 16 + nlo;
    v8f r = vzero(), z = vzero();
    float hnew[8];
#pragma unroll
    for (int g = 0; g < 3; ++g) {
      int tg = g * 16 + ct;  // n-tile index within 768-wide gate output
      v8f ai = wmma_acc<NKSI>(axi, pw_i + tg * NKSI * 256, lane, vzero());
      v8f ah = wmma_acc<NKSH>(axh, pw_h + tg * NKSH * 256, lane, vzero());
      float bi = b_i[g * HDIM + col], bh = b_h[g * HDIM + col];
      if (g == 0) {
#pragma unroll
        for (int e = 0; e < 8; ++e) r[e] = sigm(ai[e] + bi + ah[e] + bh);
      } else if (g == 1) {
#pragma unroll
        for (int e = 0; e < 8; ++e) z[e] = sigm(ai[e] + bi + ah[e] + bh);
      } else {
#pragma unroll
        for (int e = 0; e < 8; ++e) {
          float n  = fast_tanh(ai[e] + bi + r[e] * (ah[e] + bh));
          float ho = sHold[(e + 8 * half) * HDIM + col];
          hnew[e]  = (1.0f - z[e]) * n + z[e] * ho;
        }
      }
    }
#pragma unroll
    for (int e = 0; e < 8; ++e) sOut[(e + 8 * half) * HDIM + col] = hnew[e];
  }
}

// --------------------------- main persistent kernel ------------------------
__global__ __launch_bounds__(256, 1) void knet_main(
    const float* __restrict__ x_in,
    const float* __restrict__ bf,   const float* __restrict__ b_ih,
    const float* __restrict__ b_hh, const float* __restrict__ ln_g,
    const float* __restrict__ ln_b, const float* __restrict__ bg,
    const float* __restrict__ Wrp,  const float* __restrict__ brp,
    const float* __restrict__ b_ih_r, const float* __restrict__ b_hh_r,
    const float* __restrict__ Wrpr, const float* __restrict__ brpr,
    const unsigned* __restrict__ panels, float* __restrict__ out) {
  __shared__ unsigned sAx[2560];   // A-frags, up to K=320 (10 ksteps)
  __shared__ unsigned sAh[2048];   // A-frags for h (K=256)
  __shared__ float    sH[ROWS * HDIM];      // hidden state fp32
  __shared__ float    sStage[ROWS * HDIM];  // zf / h_new / K staging
  __shared__ float    sX[ROWS * DDIM];      // x_post (encoder) / curr (rollout)
  __shared__ float    sUpd[ROWS * DDIM];    // posterior update
  __shared__ float    sRP[ROWS * 2];        // rho/phi pre-activations

  const int tid  = threadIdx.x;
  const int wave = tid >> 5, lane = tid & 31;
  const int half = lane >> 4, nlo = lane & 15;
  const int b0   = blockIdx.x * ROWS;

  // -------- init: h = 0, upd = 0, x_post = y0 --------
  for (int i = tid; i < ROWS * HDIM; i += 256) sH[i] = 0.0f;
  for (int i = tid; i < ROWS * DDIM; i += 256) {
    int M = i >> 6, j = i & 63;
    sX[i]   = x_in[((size_t)(b0 + M) * QLEN) * DDIM + j];
    sUpd[i] = 0.0f;
  }
  __syncthreads();

  // ============================ encoder scan ===========================
  for (int t = 0; t < QLEN; ++t) {
    const int tp = (t == 0) ? 0 : t - 1;
    // Per-iteration opaque offset: B-frag loads stay in-loop as global loads.
    const unsigned* pan = panels + opaque_zero();

    // 1) feat = [y_t - y_prev, y_t - x_post, prev_upd]  (K=192) -> A-frags
    fill_afrag<6>(sAx, tid, [&](int M, int k) -> float {
      const float* yr = x_in + ((size_t)(b0 + M) * QLEN + t) * DDIM;
      if (k < 64)  return yr[k] - x_in[((size_t)(b0 + M) * QLEN + tp) * DDIM + k];
      if (k < 128) return yr[k - 64] - sX[M * DDIM + (k - 64)];
      return sUpd[M * DDIM + (k - 128)];
    });
    __syncthreads();

    // 2) zf = tanh(feat @ Wf^T + bf)   (N=256, K=192)
#pragma unroll
    for (int c = 0; c < 2; ++c) {
      int ct   = wave * 2 + c;
      v8f acc  = wmma_acc<6>(sAx, pan + OFF_WF + ct * 6 * 256, lane, vzero());
      float bb = bf[ct * 16 + nlo];
#pragma unroll
      for (int e = 0; e < 8; ++e)
        sStage[(e + 8 * half) * HDIM + ct * 16 + nlo] = fast_tanh(acc[e] + bb);
    }
    __syncthreads();

    // 3) A-frags for zf and h
    fill_afrag<8>(sAx, tid, [&](int M, int k) -> float { return sStage[M * HDIM + k]; });
    fill_afrag<8>(sAh, tid, [&](int M, int k) -> float { return sH[M * HDIM + k]; });
    __syncthreads();

    // 4) GRU cell -> h_new in sStage
    gru_block<8, 8>(sAx, pan + OFF_WIH, sAh, pan + OFF_WHH,
                    b_ih, b_hh, sH, sStage, wave, lane);
    __syncthreads();

    // 5) layernorm(sStage) -> sH  (16 lanes per row, shuffle reduce)
    {
      int rr = tid >> 4, sub = tid & 15;
      float s = 0.0f;
#pragma unroll
      for (int j = 0; j < 16; ++j) s += sStage[rr * HDIM + sub * 16 + j];
#pragma unroll
      for (int o = 1; o < 16; o <<= 1) s += __shfl_xor(s, o, 32);
      float mu = s * (1.0f / HDIM);
      float vv = 0.0f;
#pragma unroll
      for (int j = 0; j < 16; ++j) {
        float d = sStage[rr * HDIM + sub * 16 + j] - mu;
        vv += d * d;
      }
#pragma unroll
      for (int o = 1; o < 16; o <<= 1) vv += __shfl_xor(vv, o, 32);
      float inv = fast_rsq(vv * (1.0f / HDIM) + 1e-5f);
#pragma unroll
      for (int j = 0; j < 16; ++j) {
        int ci = sub * 16 + j;
        sH[rr * HDIM + ci] =
            (sStage[rr * HDIM + ci] - mu) * inv * ln_g[ci] + ln_b[ci];
      }
    }
    __syncthreads();

    // 6) rp = h @ Wrp^T + brp  (16x2, VALU)
    if (tid < 32) {
      int rr = tid & 15, cc = tid >> 4;
      float s = brp[cc];
      for (int k = 0; k < HDIM; ++k) s += sH[rr * HDIM + k] * Wrp[cc * HDIM + k];
      sRP[rr * 2 + cc] = s;
    }
    __syncthreads();

    // 7) K = sigmoid(h @ Wg^T + bg)  (N=64, K=256) -> sStage[0..1023]
    fill_afrag<8>(sAh, tid, [&](int M, int k) -> float { return sH[M * HDIM + k]; });
    __syncthreads();
    if (wave < 4) {
      v8f acc  = wmma_acc<8>(sAh, pan + OFF_WG + wave * 8 * 256, lane, vzero());
      float bb = bg[wave * 16 + nlo];
#pragma unroll
      for (int e = 0; e < 8; ++e)
        sStage[(e + 8 * half) * DDIM + wave * 16 + nlo] = sigm(acc[e] + bb);
    }
    __syncthreads();

    // 8) x_pri = rotate(x_post); x_post = x_pri + K*(y_t - x_pri); upd = diff
    {
#pragma unroll
      for (int pi = 0; pi < 2; ++pi) {
        int pp = tid * 2 + pi, rr = pp >> 5, j = pp & 31;
        float rho = sigm(sRP[rr * 2 + 0]) * 1.25f;
        float phi = PI_F * fast_tanh(sRP[rr * 2 + 1]);
        float cc = __cosf(phi), ss = __sinf(phi);
        float re = sX[rr * DDIM + j], im = sX[rr * DDIM + 32 + j];
        float pr  = rho * (cc * re - ss * im);
        float pim = rho * (ss * re + cc * im);
        const float* yr = x_in + ((size_t)(b0 + rr) * QLEN + t) * DDIM;
        float Kr = sStage[rr * DDIM + j], Ki = sStage[rr * DDIM + 32 + j];
        float nr = pr + Kr * (yr[j] - pr);
        float ni = pim + Ki * (yr[32 + j] - pim);
        sX[rr * DDIM + j] = nr;        sX[rr * DDIM + 32 + j] = ni;
        sUpd[rr * DDIM + j] = nr - pr; sUpd[rr * DDIM + 32 + j] = ni - pim;
      }
    }
    __syncthreads();
  }

  // ============================ rollout ================================
  for (int s = 0; s < WOUT; ++s) {
    const unsigned* pan = panels + opaque_zero();

    // A-frags: concat(h_r, curr)  K=320  and h_r  K=256
    fill_afrag<10>(sAx, tid, [&](int M, int k) -> float {
      return (k < HDIM) ? sH[M * HDIM + k] : sX[M * DDIM + (k - HDIM)];
    });
    fill_afrag<8>(sAh, tid, [&](int M, int k) -> float { return sH[M * HDIM + k]; });
    __syncthreads();

    gru_block<10, 8>(sAx, pan + OFF_WIHR, sAh, pan + OFF_WHHR,
                     b_ih_r, b_hh_r, sH, sStage, wave, lane);
    __syncthreads();

    // layernorm -> sH
    {
      int rr = tid >> 4, sub = tid & 15;
      float ssum = 0.0f;
#pragma unroll
      for (int j = 0; j < 16; ++j) ssum += sStage[rr * HDIM + sub * 16 + j];
#pragma unroll
      for (int o = 1; o < 16; o <<= 1) ssum += __shfl_xor(ssum, o, 32);
      float mu = ssum * (1.0f / HDIM);
      float vv = 0.0f;
#pragma unroll
      for (int j = 0; j < 16; ++j) {
        float d = sStage[rr * HDIM + sub * 16 + j] - mu;
        vv += d * d;
      }
#pragma unroll
      for (int o = 1; o < 16; o <<= 1) vv += __shfl_xor(vv, o, 32);
      float inv = fast_rsq(vv * (1.0f / HDIM) + 1e-5f);
#pragma unroll
      for (int j = 0; j < 16; ++j) {
        int ci = sub * 16 + j;
        sH[rr * HDIM + ci] =
            (sStage[rr * HDIM + ci] - mu) * inv * ln_g[ci] + ln_b[ci];
      }
    }
    __syncthreads();

    // rp = h_r @ Wrpr^T + brpr
    if (tid < 32) {
      int rr = tid & 15, cc = tid >> 4;
      float acc = brpr[cc];
      for (int k = 0; k < HDIM; ++k) acc += sH[rr * HDIM + k] * Wrpr[cc * HDIM + k];
      sRP[rr * 2 + cc] = acc;
    }
    __syncthreads();

    // curr = rotate(curr); emit prediction
    {
#pragma unroll
      for (int pi = 0; pi < 2; ++pi) {
        int pp = tid * 2 + pi, rr = pp >> 5, j = pp & 31;
        float rho = sigm(sRP[rr * 2 + 0]) * 1.25f;
        float phi = PI_F * fast_tanh(sRP[rr * 2 + 1]);
        float cc = __cosf(phi), sn = __sinf(phi);
        float re = sX[rr * DDIM + j], im = sX[rr * DDIM + 32 + j];
        float nr = rho * (cc * re - sn * im);
        float ni = rho * (sn * re + cc * im);
        sX[rr * DDIM + j] = nr; sX[rr * DDIM + 32 + j] = ni;
        size_t ob = ((size_t)(b0 + rr) * WOUT + s) * DDIM;
        out[ob + j]      = nr;
        out[ob + 32 + j] = ni;
      }
    }
    __syncthreads();
  }
}

// --------------------------- host entry ------------------------------------
extern "C" void kernel_launch(void* const* d_in, const int* in_sizes, int n_in,
                              void* d_out, int out_size, void* d_ws, size_t ws_size,
                              hipStream_t stream) {
  (void)in_sizes; (void)n_in; (void)out_size; (void)ws_size;
  const float* x_in   = (const float*)d_in[0];
  // d_in[1] = w_out scalar (== 32, compiled in)
  const float* Wf     = (const float*)d_in[2];
  const float* bf     = (const float*)d_in[3];
  const float* W_ih   = (const float*)d_in[4];
  const float* W_hh   = (const float*)d_in[5];
  const float* b_ih   = (const float*)d_in[6];
  const float* b_hh   = (const float*)d_in[7];
  const float* ln_g   = (const float*)d_in[8];
  const float* ln_b   = (const float*)d_in[9];
  const float* Wg     = (const float*)d_in[10];
  const float* bg     = (const float*)d_in[11];
  const float* Wrp    = (const float*)d_in[12];
  const float* brp    = (const float*)d_in[13];
  const float* W_ih_r = (const float*)d_in[14];
  const float* W_hh_r = (const float*)d_in[15];
  const float* b_ih_r = (const float*)d_in[16];
  const float* b_hh_r = (const float*)d_in[17];
  const float* Wrpr   = (const float*)d_in[18];
  const float* brpr   = (const float*)d_in[19];
  unsigned* panels = (unsigned*)d_ws;
  float* out = (float*)d_out;

  prep_panels<<<(PANEL_TOTAL + 255) / 256, 256, 0, stream>>>(
      Wf, W_ih, W_hh, Wg, W_ih_r, W_hh_r, panels);
  knet_main<<<B_SZ / ROWS, 256, 0, stream>>>(
      x_in, bf, b_ih, b_hh, ln_g, ln_b, bg, Wrp, brp,
      b_ih_r, b_hh_r, Wrpr, brpr, panels, out);
}
